// BottleneckApprox_66030827209194
// MI455X (gfx1250) — matrix operand order repaired and checked by reading, not compiled, not run on hardware
//
#include <hip/hip_runtime.h>
#include <hip/hip_bf16.h>

typedef __attribute__((ext_vector_type(16))) _Float16 v16h;
typedef __attribute__((ext_vector_type(8)))  _Float16 v8h;
typedef __attribute__((ext_vector_type(8)))  float    v8f;

typedef unsigned int u32;
typedef __attribute__((ext_vector_type(4))) u32 u32x4;
typedef __attribute__((ext_vector_type(8))) int i32x8;
typedef __attribute__((ext_vector_type(4))) int i32x4;

#define EPS 1e-5f

// ---------------------------------------------------------------------------
// TDM: 1-D tensor load (nhalves f16 elements, linear) from global to LDS.
// D# built per CDNA5 ISA ch.8: group0 = {count, lds_addr, global_addr, type=2},
// group1 = {data_size=2B, tensor_dim0/1, tile_dim0/1, stride0}.
// clang-23 toolchain: 6-arg builtin (g0, g1, g2, g3, g4, cpol).
// ---------------------------------------------------------------------------
__device__ __forceinline__ void tdm_load_1d(void* lds_ptr, const void* gptr,
                                            unsigned nhalves) {
  unsigned long long ga = (unsigned long long)(uintptr_t)gptr;
  unsigned ldsoff = (unsigned)(uintptr_t)lds_ptr;  // low 32 bits = LDS offset
  u32x4 g0;
  g0[0] = 1u;                                   // count=1, user descriptor
  g0[1] = ldsoff;                               // lds_addr
  g0[2] = (u32)ga;                              // global_addr[31:0]
  g0[3] = (u32)((ga >> 32) & 0x01FFFFFFu) | (2u << 30);  // addr[56:32] | type=2
  i32x8 g1;
  g1[0] = (int)(1u << 16);                      // data_size=1 -> 2 bytes
  g1[1] = (int)((nhalves & 0xFFFFu) << 16);     // tensor_dim0 lo16
  g1[2] = (int)(((nhalves >> 16) & 0xFFFFu) | (1u << 16)); // dim0 hi | dim1=1
  g1[3] = (int)((nhalves & 0xFFFFu) << 16);     // tile_dim0 = nhalves
  g1[4] = 1;                                    // tile_dim1 = 1
  g1[5] = (int)nhalves;                         // tensor_dim0_stride
  g1[6] = 0;
  g1[7] = 0;
  i32x4 z4 = {};
  i32x8 z8 = {};
  __builtin_amdgcn_tensor_load_to_lds(g0, g1, z4, z4, z8, 0);
}

// ---------------------------------------------------------------------------
// Prep: fold BN1 -> (a,b) table; fold BN2 into conv1 weights + bias;
// emit f16 weights pre-swizzled into WMMA B-fragment order:
//   B frag (16x16x32 f16): lane 0-15 holds K=0..7,16..23 (elem e -> K=(e&7)+16*(e>>3)),
//   lane 16-31 holds K=8..15,24..31; column N = lane&15.
// w1s layout: [ks(8)][nt(8)][lane(32)][e(16)]   (K = ks*32 + ...)
// w2s layout: [kt(36)][nt(2)][lane(32)][e(16)]  (kt -> tap=kt>>2, cgrp=(kt&3)*32)
// ---------------------------------------------------------------------------
__global__ void prep_kernel(const float* __restrict__ g1, const float* __restrict__ b1,
                            const float* __restrict__ m1, const float* __restrict__ v1,
                            const float* __restrict__ w1, const float* __restrict__ c1b,
                            const float* __restrict__ g2, const float* __restrict__ b2,
                            const float* __restrict__ m2, const float* __restrict__ v2,
                            const float* __restrict__ w2,
                            float2* __restrict__ bn1, float* __restrict__ bias1,
                            _Float16* __restrict__ w1s, _Float16* __restrict__ w2s) {
  __shared__ float s_inv2[128];
  const int t = threadIdx.x;  // 256 threads, one block
  if (t < 256) {
    float a = g1[t] * rsqrtf(v1[t] + EPS);
    bn1[t] = make_float2(a, b1[t] - m1[t] * a);
  }
  if (t < 128) {
    float iv = g2[t] * rsqrtf(v2[t] + EPS);
    s_inv2[t] = iv;
    bias1[t] = iv * c1b[t] + b2[t] - m2[t] * iv;
  }
  __syncthreads();
  for (int idx = t; idx < 32768; idx += 256) {
    int e = idx & 15, lane = (idx >> 4) & 31, nt = (idx >> 9) & 7, ks = idx >> 12;
    int o  = nt * 16 + (lane & 15);
    int kk = ks * 32 + ((lane >> 4) << 3) + (e & 7) + ((e >> 3) << 4);
    w1s[idx] = (_Float16)(s_inv2[o] * w1[o * 256 + kk]);
  }
  for (int idx = t; idx < 36864; idx += 256) {
    int e = idx & 15, lane = (idx >> 4) & 31, nt = (idx >> 9) & 1, kt = idx >> 10;
    int o   = nt * 16 + (lane & 15);
    int kin = ((lane >> 4) << 3) + (e & 7) + ((e >> 3) << 4);
    int tap = kt >> 2;
    int ci  = (kt & 3) * 32 + kin;
    w2s[idx] = (_Float16)w2[(o * 128 + ci) * 9 + tap];
  }
}

// ---------------------------------------------------------------------------
// conv1 (1x1) fused with BN1+ReLU (on the fly) and BN2+ReLU (folded):
// wave = 16 pixels x 128 out-channels, K=256 in 8 steps of 32.
// A frag (16x16x32 f16): lane half h holds K = {h*8..h*8+7, h*8+16..h*8+23}.
// ---------------------------------------------------------------------------
__global__ __launch_bounds__(128, 4)
void conv1_kernel(const float* __restrict__ x, const float2* __restrict__ gbn,
                  const _Float16* __restrict__ gw1, const float* __restrict__ gbias,
                  _Float16* __restrict__ t2) {
  __shared__ _Float16 sW[32768];
  __shared__ float2   sBN[256];

  const int tid = threadIdx.x;
  for (int i = tid; i < 256; i += 128) sBN[i] = gbn[i];
  if (tid < 32) {
    tdm_load_1d(sW, gw1, 32768);
    __builtin_amdgcn_s_wait_tensorcnt(0);
  }
  __syncthreads();

  const int wid   = blockIdx.x * 4 + (tid >> 5);
  const int lane  = tid & 31;
  const int half  = lane >> 4;
  const int m     = lane & 15;
  const int n     = wid / 196;
  const int pbase = (wid % 196) * 16;
  const int p     = pbase + m;

  v8f acc[8] = {};

  const float* xb = x + ((size_t)n * 256 + half * 8) * 3136 + p;

#pragma unroll
  for (int ks = 0; ks < 8; ++ks) {
    union { v16h v; _Float16 e[16]; } A;
    const float* xk = xb + (size_t)ks * 32 * 3136;
#pragma unroll
    for (int e = 0; e < 16; ++e) {
      const int off = (e & 7) + ((e >> 3) << 4);   // {0..7,16..23}
      float  xv = xk[(size_t)off * 3136];
      float2 ab = sBN[ks * 32 + half * 8 + off];
      A.e[e] = (_Float16)fmaxf(fmaf(xv, ab.x, ab.y), 0.0f);
    }
#pragma unroll
    for (int nt = 0; nt < 8; ++nt) {
      union { v16h v; v8h h2[2]; } B;
      const _Float16* wp = sW + ((ks * 8 + nt) * 32 + lane) * 16;
      B.h2[0] = *(const v8h*)wp;
      B.h2[1] = *(const v8h*)(wp + 8);
      acc[nt] = __builtin_amdgcn_wmma_f32_16x16x32_f16(
          false, A.v, false, B.v, (short)0, acc[nt], false, false);
    }
  }

  // Epilogue: folded BN2 bias + ReLU, store t2 as NHWC f16 (C contiguous)
  _Float16* t2b = t2 + ((size_t)n * 3136 + pbase) * 128;
#pragma unroll
  for (int nt = 0; nt < 8; ++nt) {
    const int   c    = nt * 16 + m;
    const float bias = gbias[c];
#pragma unroll
    for (int r = 0; r < 8; ++r) {
      const int pix = r + half * 8;  // C/D layout: M = r + half*8, N = lane&15
      t2b[(size_t)pix * 128 + c] = (_Float16)fmaxf(acc[nt][r] + bias, 0.0f);
    }
  }
}

// ---------------------------------------------------------------------------
// conv2 (3x3, pad 1, 128->32): implicit GEMM, K = 9 taps * 128 ch = 36 steps.
// Reads t2 NHWC f16; per-lane zero padding at borders; writes NCHW f32 slice
// of d_out at channel offset 256.
// ---------------------------------------------------------------------------
__global__ __launch_bounds__(256, 2)
void conv2_kernel(const _Float16* __restrict__ t2, const _Float16* __restrict__ gw2,
                  const float* __restrict__ c2b, float* __restrict__ out) {
  __shared__ _Float16 sW[36864];

  const int tid = threadIdx.x;
  if (tid < 32) {
    tdm_load_1d(sW, gw2, 36864);
    __builtin_amdgcn_s_wait_tensorcnt(0);
  }
  __syncthreads();

  const int wid   = blockIdx.x * 8 + (tid >> 5);
  const int lane  = tid & 31;
  const int half  = lane >> 4;
  const int m     = lane & 15;
  const int n     = wid / 196;
  const int pbase = (wid % 196) * 16;
  const int p     = pbase + m;
  const int h     = p / 56, w = p % 56;

  v8f acc[2] = {};
  const _Float16* tb = t2 + (size_t)n * 3136 * 128;

#pragma unroll
  for (int tap = 0; tap < 9; ++tap) {
    const int  dy = tap / 3 - 1, dx = tap % 3 - 1;
    const int  hh = h + dy, ww = w + dx;
    const bool valid = ((unsigned)hh < 56u) && ((unsigned)ww < 56u);
    const _Float16* src = tb + (size_t)(hh * 56 + ww) * 128 + half * 8;
#pragma unroll
    for (int g = 0; g < 4; ++g) {
      union { v16h v; v8h h2[2]; } A;
      A.h2[0] = (v8h){};
      A.h2[1] = (v8h){};
      if (valid) {
        A.h2[0] = *(const v8h*)(src + g * 32);
        A.h2[1] = *(const v8h*)(src + g * 32 + 16);
      }
      const int kt = tap * 4 + g;
#pragma unroll
      for (int nt = 0; nt < 2; ++nt) {
        union { v16h v; v8h h2[2]; } B;
        const _Float16* wp = sW + ((kt * 2 + nt) * 32 + lane) * 16;
        B.h2[0] = *(const v8h*)wp;
        B.h2[1] = *(const v8h*)(wp + 8);
        acc[nt] = __builtin_amdgcn_wmma_f32_16x16x32_f16(
            false, A.v, false, B.v, (short)0, acc[nt], false, false);
      }
    }
  }

  float* ob = out + ((size_t)n * 288 + 256) * 3136 + pbase;
#pragma unroll
  for (int nt = 0; nt < 2; ++nt) {
    const int   c    = nt * 16 + m;
    const float bias = c2b[c];
#pragma unroll
    for (int r = 0; r < 8; ++r) {
      const int pix = r + half * 8;
      ob[(size_t)c * 3136 + pix] = acc[nt][r] + bias;
    }
  }
}

// ---------------------------------------------------------------------------
// Concat: copy x (N,256,56,56) into out channels 0..255 (vectorized float4)
// ---------------------------------------------------------------------------
__global__ void copy_x_kernel(const float4* __restrict__ x, float4* __restrict__ out) {
  size_t idx = (size_t)blockIdx.x * blockDim.x + threadIdx.x;  // 6,422,528 total
  const size_t per = 200704;  // 256*3136/4 float4 per image
  size_t n = idx / per, r = idx - n * per;
  out[n * 225792 + r] = x[idx];  // 225792 = 288*3136/4
}

// ---------------------------------------------------------------------------
extern "C" void kernel_launch(void* const* d_in, const int* in_sizes, int n_in,
                              void* d_out, int out_size, void* d_ws, size_t ws_size,
                              hipStream_t stream) {
  const float* x    = (const float*)d_in[0];
  const float* g1   = (const float*)d_in[1];
  const float* b1   = (const float*)d_in[2];
  const float* m1   = (const float*)d_in[3];
  const float* v1   = (const float*)d_in[4];
  const float* w1   = (const float*)d_in[5];
  const float* c1b  = (const float*)d_in[6];
  const float* g2   = (const float*)d_in[7];
  const float* b2   = (const float*)d_in[8];
  const float* m2   = (const float*)d_in[9];
  const float* v2   = (const float*)d_in[10];
  const float* w2   = (const float*)d_in[11];
  const float* c2b  = (const float*)d_in[12];
  float* out = (float*)d_out;

  char* ws = (char*)d_ws;
  float2*   bn1   = (float2*)(ws + 0);          // 256 * 8B   = 2048
  float*    bias1 = (float*)(ws + 2048);        // 128 * 4B   = 512
  _Float16* w1s   = (_Float16*)(ws + 2560);     // 32768 * 2B = 65536
  _Float16* w2s   = (_Float16*)(ws + 68096);    // 36864 * 2B = 73728
  _Float16* t2    = (_Float16*)(ws + 141824);   // 32*3136*128 f16 = 25.7 MB

  prep_kernel<<<1, 256, 0, stream>>>(g1, b1, m1, v1, w1, c1b,
                                     g2, b2, m2, v2, w2,
                                     bn1, bias1, w1s, w2s);

  // conv1: 32 images * 196 tiles(16 px) = 6272 waves, 4 waves/block
  conv1_kernel<<<1568, 128, 0, stream>>>(x, bn1, w1s, bias1, t2);

  // conv2: 6272 waves, 8 waves/block
  conv2_kernel<<<784, 256, 0, stream>>>(t2, w2s, c2b, out);

  // concat copy of x into channels 0..255
  copy_x_kernel<<<25088, 256, 0, stream>>>((const float4*)x, (float4*)out);
}
